// PointerBasedCrossModalModule_71330816852276
// MI455X (gfx1250) — matrix operand order repaired and checked by reading, not compiled, not run on hardware
//
#include <hip/hip_runtime.h>
#include <math.h>

#define NP_ 2000
#define NV_ 100000
#define E_  1000000
#define H_  128
#define PITCH 272   // staged LDS row pitch in bytes: 256 data + 16 pad (bank-conflict avoidance)

typedef __attribute__((ext_vector_type(16))) __bf16 v16bf;
typedef __attribute__((ext_vector_type(8)))  __bf16 v8bf;
typedef __attribute__((ext_vector_type(8)))  float  v8f;
typedef __attribute__((ext_vector_type(4)))  float  v4f;
typedef __attribute__((ext_vector_type(4)))  unsigned u32x4;
typedef __attribute__((ext_vector_type(8)))  int      i32x8;
typedef __attribute__((ext_vector_type(4)))  int      i32x4;

#if defined(__has_builtin)
#  if __has_builtin(__builtin_amdgcn_tensor_load_to_lds) && __has_builtin(__builtin_amdgcn_s_wait_tensorcnt)
#    define USE_TDM 1
#  endif
#endif
#ifndef USE_TDM
#  define USE_TDM 0
#endif

// ---------- helpers ----------
__device__ inline __bf16 f2bf(float f) { return (__bf16)f; }   // native v_cvt on gfx1250

__device__ inline v8f zero8() {
  v8f r;
#pragma unroll
  for (int i = 0; i < 8; ++i) r[i] = 0.0f;
  return r;
}

__device__ inline float gumbelf(float u) {
  u = fminf(fmaxf(u, 1e-9f), 1.0f - 1e-9f);
  return -logf(-logf(u));
}

// ---------- TDM stage: WT (128x128 bf16, contiguous) -> LDS with 16B row padding ----------
__device__ inline void stage_WT(const __bf16* WT, char* sWT) {
#if USE_TDM
  if (threadIdx.x < 32) {     // one wave issues the tensor DMA; TENSORcnt is per-wave
    unsigned lds = (unsigned)(size_t)sWT;                 // flat LDS addr low 32b == LDS offset
    unsigned long long ga = (unsigned long long)(size_t)WT;
    // D# group0: count=1 | lds_addr | global_addr(57b) | type=2
    u32x4 g0 = { 1u, lds, (unsigned)ga,
                 (unsigned)((ga >> 32) & 0x01FFFFFFu) | (2u << 30) };
    // D# group1: data_size=2B, pad_enable, pad_interval=64 dwords (code 5), pad_amount=4 dwords (code 3)
    i32x8 g1;
    g1[0] = (int)((1u << 16) | (1u << 20) | (5u << 22) | (3u << 25));
    g1[1] = (int)((16384u & 0xFFFFu) << 16);              // tensor_dim0[15:0] -> bits 63:48
    g1[2] = (int)(((16384u >> 16) & 0xFFFFu) | (1u << 16)); // dim0[31:16]; tensor_dim1=1
    g1[3] = (int)((16384u & 0xFFFFu) << 16);              // tile_dim0 = 16384 elements
    g1[4] = 0;                                            // tile_dim1 = tile_dim2 = 0 (unused)
    g1[5] = 16384;                                        // tensor_dim0_stride
    g1[6] = 0; g1[7] = 0;
    i32x4 gz4 = { 0, 0, 0, 0 };
    i32x8 gz8 = { 0, 0, 0, 0, 0, 0, 0, 0 };
    __builtin_amdgcn_tensor_load_to_lds(g0, g1, gz4, gz4, gz8, 0);  // 6-arg form (clang-23)
    __builtin_amdgcn_s_wait_tensorcnt((short)0);
  }
  __syncthreads();
#else
  for (int i = threadIdx.x; i < 2048; i += 256) {         // 2048 x 16B chunks
    int row = i >> 4, inrow = (i & 15) * 16;
    *(v4f*)(sWT + row * PITCH + inrow) = *(const v4f*)((const char*)WT + i * 16);
  }
  __syncthreads();
#endif
}

// B fragment for (ncol, kc, half) from staged LDS (two aligned 16B ds loads)
__device__ inline v16bf load_bfrag(const char* sWT, int ncol, int kc, int half) {
  const char* p = sWT + ncol * PITCH + (kc + half * 8) * 2;
  union { v16bf v; v8bf h[2]; } ub;
  ub.h[0] = *(const v8bf*)p;
  ub.h[1] = *(const v8bf*)(p + 32);
  return ub.v;
}

// ---------- prep: transpose-convert W_p,W_v,W_m1 (fp32 HxH) -> bf16 WT (n-major) ----------
__global__ void prep_weights_kernel(const float* __restrict__ Wp, const float* __restrict__ Wv,
                                    const float* __restrict__ Wm1,
                                    __bf16* __restrict__ WTp, __bf16* __restrict__ WTv,
                                    __bf16* __restrict__ WTm1) {
  int tid = blockIdx.x * blockDim.x + threadIdx.x;
  if (tid >= H_ * H_) return;
  int k = tid >> 7, n = tid & 127;
  WTp [n * H_ + k] = f2bf(Wp [k * H_ + n]);
  WTv [n * H_ + k] = f2bf(Wv [k * H_ + n]);
  WTm1[n * H_ + k] = f2bf(Wm1[k * H_ + n]);
}

// ---------- init: v_out = v, zero segmax keys ----------
__global__ void init_kernel(const float* __restrict__ v, float* __restrict__ v_out,
                            unsigned* __restrict__ segmax) {
  int i = blockIdx.x * blockDim.x + threadIdx.x;          // over NV*32 float4s (exact)
  ((v4f*)v_out)[i] = ((const v4f*)v)[i];
  if (i < NV_) segmax[i] = 0u;                            // y_un > 0 -> uint-bit order valid
}

// ---------- WMMA GEMM: out[M x H] = A[M x H] @ W + b ----------
__global__ __launch_bounds__(256) void gemm_bias_kernel(const float* __restrict__ A,
                                                        const __bf16* __restrict__ WT,
                                                        const float* __restrict__ bias,
                                                        float* __restrict__ out, int M) {
  __shared__ __align__(16) char sWT[H_ * PITCH];
  stage_WT(WT, sWT);

  int wave = threadIdx.x >> 5, lane = threadIdx.x & 31;
  int row0 = (blockIdx.x * 8 + wave) * 16;
  if (row0 >= M) return;
  int half = lane >> 4, l16 = lane & 15;

  v8f acc[8];
#pragma unroll
  for (int t = 0; t < 8; ++t) acc[t] = zero8();

  int arow = row0 + l16; if (arow >= M) arow = M - 1;     // pad tile rows (stores guarded)
  const float* Ar = A + (size_t)arow * H_;

#pragma unroll
  for (int kt = 0; kt < 4; ++kt) {
    int kc = kt * 32, kb = half * 8;
    v4f a0 = *(const v4f*)(Ar + kc + kb);
    v4f a1 = *(const v4f*)(Ar + kc + kb + 4);
    v4f a2 = *(const v4f*)(Ar + kc + 16 + kb);
    v4f a3 = *(const v4f*)(Ar + kc + 16 + kb + 4);
    v16bf af;
#pragma unroll
    for (int j = 0; j < 4; ++j) {
      af[j]      = f2bf(a0[j]);
      af[4 + j]  = f2bf(a1[j]);
      af[8 + j]  = f2bf(a2[j]);
      af[12 + j] = f2bf(a3[j]);
    }
#pragma unroll
    for (int nt = 0; nt < 8; ++nt) {
      v16bf bf = load_bfrag(sWT, nt * 16 + l16, kc, half);
      acc[nt] = __builtin_amdgcn_wmma_f32_16x16x32_bf16(false, af, false, bf,
                                                        (short)0, acc[nt], false, false);
    }
  }

#pragma unroll
  for (int nt = 0; nt < 8; ++nt) {
    int ncol = nt * 16 + l16;
    float b = bias[ncol];
#pragma unroll
    for (int r = 0; r < 8; ++r) {
      int m = row0 + half * 8 + r;
      if (m < M) out[(size_t)m * H_ + ncol] = acc[nt][r] + b;
    }
  }
}

// ---------- fused mask: lrelu(v@W_m1+b1)@W_m2+b2 -> gumbel softmax2 -> soft/hard ----------
__global__ __launch_bounds__(256) void mask_kernel(const float* __restrict__ V,
                                                   const __bf16* __restrict__ WTm1,
                                                   const float* __restrict__ b_m1,
                                                   const float* __restrict__ Wm2,
                                                   const float* __restrict__ b_m2,
                                                   const float* __restrict__ u_mask,
                                                   float* __restrict__ mask_soft,
                                                   float* __restrict__ mask_hard) {
  __shared__ __align__(16) char sWT[H_ * PITCH];
  stage_WT(WTm1, sWT);

  const int M = NV_;
  int wave = threadIdx.x >> 5, lane = threadIdx.x & 31;
  int row0 = (blockIdx.x * 8 + wave) * 16;
  if (row0 >= M) return;
  int half = lane >> 4, l16 = lane & 15;

  v8f acc[8];
#pragma unroll
  for (int t = 0; t < 8; ++t) acc[t] = zero8();

  const float* Ar = V + (size_t)(row0 + l16) * H_;        // M % 16 == 0
#pragma unroll
  for (int kt = 0; kt < 4; ++kt) {
    int kc = kt * 32, kb = half * 8;
    v4f a0 = *(const v4f*)(Ar + kc + kb);
    v4f a1 = *(const v4f*)(Ar + kc + kb + 4);
    v4f a2 = *(const v4f*)(Ar + kc + 16 + kb);
    v4f a3 = *(const v4f*)(Ar + kc + 16 + kb + 4);
    v16bf af;
#pragma unroll
    for (int j = 0; j < 4; ++j) {
      af[j] = f2bf(a0[j]); af[4 + j] = f2bf(a1[j]);
      af[8 + j] = f2bf(a2[j]); af[12 + j] = f2bf(a3[j]);
    }
#pragma unroll
    for (int nt = 0; nt < 8; ++nt) {
      v16bf bf = load_bfrag(sWT, nt * 16 + l16, kc, half);
      acc[nt] = __builtin_amdgcn_wmma_f32_16x16x32_bf16(false, af, false, bf,
                                                        (short)0, acc[nt], false, false);
    }
  }

  float p0[8], p1[8];
#pragma unroll
  for (int r = 0; r < 8; ++r) { p0[r] = 0.0f; p1[r] = 0.0f; }
#pragma unroll
  for (int nt = 0; nt < 8; ++nt) {
    int ncol = nt * 16 + l16;
    float bm = b_m1[ncol];
    float w0 = Wm2[ncol * 2 + 0], w1 = Wm2[ncol * 2 + 1];
#pragma unroll
    for (int r = 0; r < 8; ++r) {
      float hv = acc[nt][r] + bm;
      hv = (hv >= 0.0f) ? hv : 0.01f * hv;                // leaky_relu slope 0.01
      p0[r] += hv * w0; p1[r] += hv * w1;
    }
  }
#pragma unroll
  for (int r = 0; r < 8; ++r) {
#pragma unroll
    for (int mk = 8; mk >= 1; mk >>= 1) {
      p0[r] += __shfl_xor(p0[r], mk, 16);
      p1[r] += __shfl_xor(p1[r], mk, 16);
    }
  }
  if (l16 == 0) {
    float bm0 = b_m2[0], bm1 = b_m2[1];
#pragma unroll
    for (int r = 0; r < 8; ++r) {
      int m = row0 + half * 8 + r;
      float z0 = p0[r] + bm0 + gumbelf(u_mask[m * 2 + 0]);  // TAU == 1
      float z1 = p1[r] + bm1 + gumbelf(u_mask[m * 2 + 1]);
      mask_soft[m] = 1.0f / (1.0f + expf(z1 - z0));
      mask_hard[m] = (z0 >= z1) ? 1.0f : 0.0f;              // argmax ties -> index 0
    }
  }
}

// ---------- per-edge score: z = theta . tanh(xp[ei0]+vv[ei1]) + gumbel; per-block max ----------
__global__ __launch_bounds__(256) void edge_logits_kernel(const int* __restrict__ ei0,
                                                          const int* __restrict__ ei1,
                                                          const float* __restrict__ xp,
                                                          const float* __restrict__ vv,
                                                          const float* __restrict__ theta,
                                                          const float* __restrict__ u_edge,
                                                          float* __restrict__ z,
                                                          float* __restrict__ zpart) {
  __shared__ float smax[8];
  int wave = threadIdx.x >> 5, lane = threadIdx.x & 31;
  int e = blockIdx.x * 8 + wave;                          // grid exact: E_ % 8 == 0
  float zz = -3.0e38f;
  if (e < E_) {
    int p = ei0[e], n = ei1[e];
    v4f xa = *(const v4f*)(xp + (size_t)p * H_ + lane * 4);
    v4f va = *(const v4f*)(vv + (size_t)n * H_ + lane * 4);
    v4f th = *(const v4f*)(theta + lane * 4);
    float s = 0.0f;
#pragma unroll
    for (int j = 0; j < 4; ++j) s += tanhf(xa[j] + va[j]) * th[j];
#pragma unroll
    for (int mk = 16; mk >= 1; mk >>= 1) s += __shfl_xor(s, mk, 32);
    zz = s + gumbelf(u_edge[e]);                          // TAU == 1
    if (lane == 0) z[e] = zz;
  }
  if (lane == 0) smax[wave] = zz;
  __syncthreads();
  if (threadIdx.x == 0) {
    float m = smax[0];
#pragma unroll
    for (int i = 1; i < 8; ++i) m = fmaxf(m, smax[i]);
    zpart[blockIdx.x] = m;                                // no single-address atomic hotspot
  }
}

// ---------- deterministic max over block partials ----------
__global__ __launch_bounds__(256) void max_partials_kernel(const float* __restrict__ zp, int n,
                                                           float* __restrict__ zmax) {
  __shared__ float sd[256];
  float m = -3.0e38f;
  for (int i = threadIdx.x; i < n; i += 256) m = fmaxf(m, zp[i]);
  sd[threadIdx.x] = m;
  __syncthreads();
  for (int s = 128; s > 0; s >>= 1) {
    if (threadIdx.x < s) sd[threadIdx.x] = fmaxf(sd[threadIdx.x], sd[threadIdx.x + s]);
    __syncthreads();
  }
  if (threadIdx.x == 0) *zmax = sd[0];
}

// ---------- exp(z - zmax), per-node max (bit atomics), per-block partial sums ----------
__global__ __launch_bounds__(256) void edge_exp_kernel(const float* __restrict__ z,
                                                       const float* __restrict__ zmax_p,
                                                       const int* __restrict__ ei1,
                                                       float* __restrict__ y_un,
                                                       unsigned* __restrict__ segmax,
                                                       float* __restrict__ partials) {
  __shared__ float sdata[256];
  float zmax = *zmax_p;
  float local = 0.0f;
  for (int e = blockIdx.x * blockDim.x + threadIdx.x; e < E_; e += gridDim.x * blockDim.x) {
    float yu = expf(z[e] - zmax);                         // in (0, 1]
    y_un[e] = yu;
    atomicMax(&segmax[ei1[e]], __float_as_uint(yu));      // positive: uint order == float order
    local += yu;
  }
  sdata[threadIdx.x] = local;
  __syncthreads();
  for (int s = 128; s > 0; s >>= 1) {
    if (threadIdx.x < s) sdata[threadIdx.x] += sdata[threadIdx.x + s];
    __syncthreads();
  }
  if (threadIdx.x == 0) partials[blockIdx.x] = sdata[0];
}

// ---------- deterministic final sum -> 1/total ----------
__global__ __launch_bounds__(256) void sum_partials_kernel(const float* __restrict__ partials,
                                                           int n, float* __restrict__ inv_total) {
  __shared__ float sdata[256];
  float local = 0.0f;
  for (int i = threadIdx.x; i < n; i += 256) local += partials[i];
  sdata[threadIdx.x] = local;
  __syncthreads();
  for (int s = 128; s > 0; s >>= 1) {
    if (threadIdx.x < s) sdata[threadIdx.x] += sdata[threadIdx.x + s];
    __syncthreads();
  }
  if (threadIdx.x == 0) *inv_total = 1.0f / sdata[0];
}

// ---------- final: attention_hard + scatter v_out += mask_soft[n]*y*x[p] ----------
__global__ __launch_bounds__(256) void edge_final_kernel(const int* __restrict__ ei0,
                                                         const int* __restrict__ ei1,
                                                         const float* __restrict__ x,
                                                         const float* __restrict__ y_un,
                                                         const unsigned* __restrict__ segmax,
                                                         const float* __restrict__ inv_total,
                                                         const float* __restrict__ mask_soft,
                                                         float* __restrict__ v_out,
                                                         float* __restrict__ att_hard) {
  int wave = threadIdx.x >> 5, lane = threadIdx.x & 31;
  int e = blockIdx.x * 8 + wave;
  if (e >= E_) return;
  int p = ei0[e], n = ei1[e];
  float yu = y_un[e];
  float y = yu * (*inv_total);
  if (lane == 0) att_hard[e] = (__float_as_uint(yu) == segmax[n]) ? 1.0f : 0.0f;
  float coeff = mask_soft[n] * y;
  v4f xa = *(const v4f*)(x + (size_t)p * H_ + lane * 4);
  float* vr = v_out + (size_t)n * H_ + lane * 4;
#pragma unroll
  for (int j = 0; j < 4; ++j) atomicAdd(vr + j, coeff * xa[j]);
}

// ---------- host launcher ----------
extern "C" void kernel_launch(void* const* d_in, const int* in_sizes, int n_in,
                              void* d_out, int out_size, void* d_ws, size_t ws_size,
                              hipStream_t stream) {
  const float* x    = (const float*)d_in[0];
  const float* v    = (const float*)d_in[1];
  const int*   cei  = (const int*)d_in[2];
  const float* u_m  = (const float*)d_in[3];
  const float* u_e  = (const float*)d_in[4];
  const float* W_p  = (const float*)d_in[5];
  const float* b_p  = (const float*)d_in[6];
  const float* W_v  = (const float*)d_in[7];
  const float* b_v  = (const float*)d_in[8];
  const float* W_m1 = (const float*)d_in[9];
  const float* b_m1 = (const float*)d_in[10];
  const float* W_m2 = (const float*)d_in[11];
  const float* b_m2 = (const float*)d_in[12];
  const float* th   = (const float*)d_in[13];
  const int* ei0 = cei;
  const int* ei1 = cei + E_;

  float* out       = (float*)d_out;
  float* v_out     = out;                                 // NV*H
  float* mask_hard = out + (size_t)NV_ * H_;              // NV
  float* att_hard  = mask_hard + NV_;                     // E

  // workspace layout (bytes); total ~61.6 MB
  char* ws = (char*)d_ws;
  __bf16*   WTp     = (__bf16*)(ws + 0);                  // 32768
  __bf16*   WTv     = (__bf16*)(ws + 32768);              // 32768
  __bf16*   WTm1    = (__bf16*)(ws + 65536);              // 32768
  float*    xp      = (float*)(ws + 98304);               // NP*H*4
  float*    vv      = (float*)(ws + 1122304);             // NV*H*4
  float*    msoft   = (float*)(ws + 52322304);            // NV*4
  float*    zbuf    = (float*)(ws + 52722304);            // E*4
  float*    y_un    = (float*)(ws + 56722304);            // E*4
  unsigned* segmax  = (unsigned*)(ws + 60722304);         // NV*4
  float*    parts   = (float*)(ws + 61122304);            // 1024*4
  float*    zpart   = (float*)(ws + 61126400);            // 125000*4
  float*    zmaxf   = (float*)(ws + 61626400);            // 4
  float*    inv_tot = (float*)(ws + 61626404);            // 4

  prep_weights_kernel<<<(H_ * H_ + 255) / 256, 256, 0, stream>>>(W_p, W_v, W_m1, WTp, WTv, WTm1);
  init_kernel<<<12500, 256, 0, stream>>>(v, v_out, segmax);
  gemm_bias_kernel<<<16, 256, 0, stream>>>(x, WTp, b_p, xp, NP_);
  gemm_bias_kernel<<<782, 256, 0, stream>>>(v, WTv, b_v, vv, NV_);
  mask_kernel<<<782, 256, 0, stream>>>(v, WTm1, b_m1, W_m2, b_m2, u_m, msoft, mask_hard);
  edge_logits_kernel<<<E_ / 8, 256, 0, stream>>>(ei0, ei1, xp, vv, th, u_e, zbuf, zpart);
  max_partials_kernel<<<1, 256, 0, stream>>>(zpart, E_ / 8, zmaxf);
  edge_exp_kernel<<<1024, 256, 0, stream>>>(zbuf, zmaxf, ei1, y_un, segmax, parts);
  sum_partials_kernel<<<1, 256, 0, stream>>>(parts, 1024, inv_tot);
  edge_final_kernel<<<E_ / 8, 256, 0, stream>>>(ei0, ei1, x, y_un, segmax, inv_tot,
                                                msoft, v_out, att_hard);
}
